// RwkvSelfAttention_14980845929068
// MI455X (gfx1250) — matrix operand order repaired and checked by reading, not compiled, not run on hardware
//
#include <hip/hip_runtime.h>

#define BQ 4
#define TT 2048
#define CCH 2048
#define HH 32
#define NNH 64
#define MMR (BQ*TT)      // 8192 rows
#define EPSV 1e-5f

typedef __attribute__((ext_vector_type(16))) __bf16 v16bf;
typedef __attribute__((ext_vector_type(8)))  float  v8f;
typedef __attribute__((ext_vector_type(4)))  int    v4i;

union Frag { v16bf v; uint4 q[2]; };

// -------- CDNA5 async global->LDS support (guarded; fallback = reg staging)
#if defined(__has_builtin)
#if __has_builtin(__builtin_amdgcn_global_load_async_to_lds_b128)
#define ASYNC_B128 1
#endif
#if __has_builtin(__builtin_amdgcn_global_load_async_to_lds_b32)
#define ASYNC_B32 1
#endif
#endif

typedef __attribute__((address_space(1))) v4i gv4i;   // global int4
typedef __attribute__((address_space(3))) v4i lv4i;   // LDS int4
typedef __attribute__((address_space(1))) int gi32;   // global int
typedef __attribute__((address_space(3))) int li32;   // LDS int

__device__ __forceinline__ void wait_async0() {
#if defined(__has_builtin) && __has_builtin(__builtin_amdgcn_s_wait_asynccnt)
  __builtin_amdgcn_s_wait_asynccnt(0);
#else
  asm volatile("s_wait_asynccnt 0x0" ::: "memory");
#endif
}

__device__ __forceinline__ unsigned short f2bf(float f) {
  unsigned int u = __float_as_uint(f);
  u += 0x7fffu + ((u >> 16) & 1u);      // round-to-nearest-even
  return (unsigned short)(u >> 16);
}

// ---------------------------------------------------------------------------
// Transpose + fp32->bf16 convert: dst[n*C+k] = bf16(src[k*C+n])
// ---------------------------------------------------------------------------
__global__ void k_transpose_bf16(const float* __restrict__ src,
                                 unsigned short* __restrict__ dst) {
  __shared__ float tile[32][33];
  const int bx = blockIdx.x * 32, by = blockIdx.y * 32;
  const int tx = threadIdx.x, ty = threadIdx.y;   // 32 x 8
#pragma unroll
  for (int i = 0; i < 32; i += 8)
    tile[ty + i][tx] = src[(size_t)(by + ty + i) * CCH + bx + tx];
  __syncthreads();
#pragma unroll
  for (int i = 0; i < 32; i += 8)
    dst[(size_t)(bx + ty + i) * CCH + by + tx] = f2bf(tile[tx][ty + i]);
}

// ---------------------------------------------------------------------------
// Time-shift mix: xq = h*mix + shifted*(1-mix), write bf16 (4 mixes fused)
// ---------------------------------------------------------------------------
#define MIX1(dstp, mixp)                                                     \
  {                                                                          \
    float4 m4 = *(const float4*)&mixp[c];                                    \
    ushort4 o;                                                               \
    o.x = f2bf(h4.x * m4.x + s4.x * (1.f - m4.x));                           \
    o.y = f2bf(h4.y * m4.y + s4.y * (1.f - m4.y));                           \
    o.z = f2bf(h4.z * m4.z + s4.z * (1.f - m4.z));                           \
    o.w = f2bf(h4.w * m4.w + s4.w * (1.f - m4.w));                           \
    *(ushort4*)&dstp[(size_t)mrow * CCH + c] = o;                            \
  }

__global__ __launch_bounds__(256) void k_mix(
    const float* __restrict__ hidden, const float* __restrict__ mr,
    const float* __restrict__ mk, const float* __restrict__ mv,
    const float* __restrict__ mg, unsigned short* __restrict__ xr,
    unsigned short* __restrict__ xk, unsigned short* __restrict__ xv,
    unsigned short* __restrict__ xg) {
  const int vid = blockIdx.x * 256 + threadIdx.x;   // MMR*CCH/4 threads
  const int mrow = vid >> 9;                        // CCH/4 = 512
  const int c = (vid & 511) << 2;
  const int t = mrow & (TT - 1);
  float4 h4 = *(const float4*)&hidden[(size_t)mrow * CCH + c];
  float4 s4 = make_float4(0.f, 0.f, 0.f, 0.f);
  if (t > 0) s4 = *(const float4*)&hidden[(size_t)(mrow - 1) * CCH + c];
  MIX1(xr, mr);
  MIX1(xk, mk);
  MIX1(xv, mv);
  MIX1(xg, mg);
}

// ---------------------------------------------------------------------------
// bf16 GEMM with v_wmma_f32_16x16x32_bf16:
//   D[M,N] (f32) = A[M,K] (bf16, row-major) * Bt[N,K]^T (bf16, N-major)
// Block tile 128x128, BK=32, 8 waves, wave tile 32x64 (2x4 WMMA tiles).
// Staging: GLOBAL_LOAD_ASYNC_TO_LDS_B128 double-buffered (ASYNCcnt),
// fallback to register staging if the builtin is unavailable.
// ---------------------------------------------------------------------------
#define BM 128
#define BN 128
#define BK 32
#define LDK 40   // padded K stride in LDS (ushorts) -> 80B rows, 16B aligned

__global__ __launch_bounds__(256) void k_gemm_bf16(
    const unsigned short* __restrict__ A, const unsigned short* __restrict__ Bt,
    float* __restrict__ D, int K, int Ntot) {
  __shared__ unsigned short sA[2][BM * LDK];
  __shared__ unsigned short sB[2][BN * LDK];
  const int tid = threadIdx.x;
  const int wid = tid >> 5, lane = tid & 31;
  const int wm = (wid & 3) * 32;     // wave M offset within block tile
  const int wn = (wid >> 2) * 64;    // wave N offset
  const int m0 = blockIdx.y * BM;
  const int n0 = blockIdx.x * BN;
  const int lrow = lane & 15, lhalf = lane >> 4;

  const v8f vzero = {0.f, 0.f, 0.f, 0.f, 0.f, 0.f, 0.f, 0.f};
  v8f acc[2][4];
#pragma unroll
  for (int i = 0; i < 2; ++i)
#pragma unroll
    for (int j = 0; j < 4; ++j) acc[i][j] = vzero;

  // per-thread staging coords: 2 chunks of 8 bf16 per tile
  const int srow0 = tid >> 2, scol = (tid & 3) * 8;
  const int srow1 = srow0 + 64;
  const int KS = K / BK;

#if ASYNC_B128
  auto stageAsync = [&](int ks, int buf) {
    __builtin_amdgcn_global_load_async_to_lds_b128(
        (gv4i*)&A[(size_t)(m0 + srow0) * K + ks * BK + scol],
        (lv4i*)&sA[buf][srow0 * LDK + scol], 0, 0);
    __builtin_amdgcn_global_load_async_to_lds_b128(
        (gv4i*)&A[(size_t)(m0 + srow1) * K + ks * BK + scol],
        (lv4i*)&sA[buf][srow1 * LDK + scol], 0, 0);
    __builtin_amdgcn_global_load_async_to_lds_b128(
        (gv4i*)&Bt[(size_t)(n0 + srow0) * K + ks * BK + scol],
        (lv4i*)&sB[buf][srow0 * LDK + scol], 0, 0);
    __builtin_amdgcn_global_load_async_to_lds_b128(
        (gv4i*)&Bt[(size_t)(n0 + srow1) * K + ks * BK + scol],
        (lv4i*)&sB[buf][srow1 * LDK + scol], 0, 0);
  };
  stageAsync(0, 0);
#else
  uint4 ra0, ra1, rb0, rb1;
  auto stageLoad = [&](int ks) {
    ra0 = *(const uint4*)&A[(size_t)(m0 + srow0) * K + ks * BK + scol];
    ra1 = *(const uint4*)&A[(size_t)(m0 + srow1) * K + ks * BK + scol];
    rb0 = *(const uint4*)&Bt[(size_t)(n0 + srow0) * K + ks * BK + scol];
    rb1 = *(const uint4*)&Bt[(size_t)(n0 + srow1) * K + ks * BK + scol];
  };
  auto stageStore = [&](int buf) {
    *(uint4*)&sA[buf][srow0 * LDK + scol] = ra0;
    *(uint4*)&sA[buf][srow1 * LDK + scol] = ra1;
    *(uint4*)&sB[buf][srow0 * LDK + scol] = rb0;
    *(uint4*)&sB[buf][srow1 * LDK + scol] = rb1;
  };
  stageLoad(0);
  stageStore(0);
#endif

  for (int ks = 0; ks < KS; ++ks) {
    const int buf = ks & 1;
#if ASYNC_B128
    wait_async0();          // our async loads for this buffer have landed
    __syncthreads();        // everyone's have
    if (ks + 1 < KS) stageAsync(ks + 1, (ks + 1) & 1);
#else
    __syncthreads();
    if (ks + 1 < KS) stageLoad(ks + 1);
#endif
    if (ks + 2 < KS) {      // gfx1250 global_prefetch_b8 two tiles ahead
      __builtin_prefetch(&A[(size_t)(m0 + (tid >> 1)) * K + (ks + 2) * BK], 0, 1);
      __builtin_prefetch(&Bt[(size_t)(n0 + (tid >> 1)) * K + (ks + 2) * BK], 0, 1);
    }
    Frag fa[2], fb[4];
#pragma unroll
    for (int im = 0; im < 2; ++im) {
      // A 16x32 bf16 layout: lane holds row=lrow; VGPR0-3: K=lhalf*8+0..7,
      // VGPR4-7: K=16+lhalf*8+0..7  -> two 16B LDS loads
      const unsigned short* p = &sA[buf][(wm + im * 16 + lrow) * LDK];
      fa[im].q[0] = *(const uint4*)(p + lhalf * 8);
      fa[im].q[1] = *(const uint4*)(p + 16 + lhalf * 8);
    }
#pragma unroll
    for (int in = 0; in < 4; ++in) {
      // B 32x16 bf16 layout: lane holds col=lrow, K = lhalf*8+0..7, 16+lhalf*8+0..7
      const unsigned short* p = &sB[buf][(wn + in * 16 + lrow) * LDK + lhalf * 8];
      fb[in].q[0] = *(const uint4*)(p);
      fb[in].q[1] = *(const uint4*)(p + 16);
    }
#pragma unroll
    for (int im = 0; im < 2; ++im)
#pragma unroll
      for (int in = 0; in < 4; ++in)
        acc[im][in] = __builtin_amdgcn_wmma_f32_16x16x32_bf16(
            false, fa[im].v, false, fb[in].v, (short)0, acc[im][in], false,
            false);
#if !ASYNC_B128
    if (ks + 1 < KS) stageStore((ks + 1) & 1);
#endif
  }

  // C/D layout: VGPR r -> M = r + lhalf*8, N = lrow
#pragma unroll
  for (int im = 0; im < 2; ++im)
#pragma unroll
    for (int in = 0; in < 4; ++in) {
      const int rbase = m0 + wm + im * 16 + lhalf * 8;
      const int ccol = n0 + wn + in * 16 + lrow;
#pragma unroll
      for (int r = 0; r < 8; ++r)
        D[(size_t)(rbase + r) * Ntot + ccol] = acc[im][in][r];
    }
}

// ---------------------------------------------------------------------------
// WKV recurrence: one block per (b,h); 256 threads: m = tid&63 (out column),
// g = tid>>6 (16-row state slice). Double-buffered LDS r/k/v via async-to-LDS
// (fallback: register round-trip), 1 barrier per timestep.
// ---------------------------------------------------------------------------
__global__ __launch_bounds__(256) void k_wkv(
    const float* __restrict__ r, const float* __restrict__ k,
    const float* __restrict__ v, const float* __restrict__ td,
    const float* __restrict__ tf, const float* __restrict__ init_state,
    float* __restrict__ wkv, float* __restrict__ fstate) {
  __shared__ float buf[2][3 * NNH];
  __shared__ float partial[2][256];
  const int bh = blockIdx.x;
  const int b = bh >> 5, h = bh & 31;
  const int tid = threadIdx.x;
  const int m = tid & 63, g = tid >> 6;
  const int n0 = g * 16;
  float S[16], u[16], w[16];
#pragma unroll
  for (int i = 0; i < 16; ++i) {
    int n = n0 + i;
    u[i] = tf[h * NNH + n];
    w[i] = __expf(-__expf(td[h * NNH + n]));
    S[i] = init_state[((size_t)(b * HH + h) * NNH + n) * NNH + m];
  }
  const size_t base = (size_t)(b * TT) * CCH + h * NNH;
  const float* src0 = tid < NNH ? r : (tid < 2 * NNH ? k : v);
  if (tid < 3 * NNH) {
#if ASYNC_B32
    __builtin_amdgcn_global_load_async_to_lds_b32(
        (gi32*)&src0[base + (tid & 63)], (li32*)&buf[0][tid], 0, 0);
#else
    buf[0][tid] = src0[base + (tid & 63)];
#endif
  }
#if ASYNC_B32
  wait_async0();
#endif
  __syncthreads();
  for (int t = 0; t < TT; ++t) {
    const int cur = t & 1;
    const bool ld = (tid < 3 * NNH) && (t + 1 < TT);
#if ASYNC_B32
    if (ld)
      __builtin_amdgcn_global_load_async_to_lds_b32(
          (gi32*)&src0[base + (size_t)(t + 1) * CCH + (tid & 63)],
          (li32*)&buf[1 - cur][tid], 0, 0);
#else
    float nxt = 0.f;
    if (ld) nxt = src0[base + (size_t)(t + 1) * CCH + (tid & 63)];
#endif
    const float vm = buf[cur][2 * NNH + m];
    float acc = 0.f;
#pragma unroll
    for (int i = 0; i < 16; ++i) {
      float rn = buf[cur][n0 + i];
      float kn = buf[cur][NNH + n0 + i];
      float kv = kn * vm;
      acc += rn * fmaf(u[i], kv, S[i]);
      S[i] = fmaf(w[i], S[i], kv);
    }
    partial[cur][tid] = acc;
#if ASYNC_B32
    wait_async0();
#else
    if (ld) buf[1 - cur][tid] = nxt;
#endif
    __syncthreads();
    if (g == 0) {
      float o = partial[cur][m] + partial[cur][64 + m] + partial[cur][128 + m] +
                partial[cur][192 + m];
      wkv[base + (size_t)t * CCH + m] = o;
    }
  }
#pragma unroll
  for (int i = 0; i < 16; ++i)
    fstate[((size_t)(b * HH + h) * NNH + n0 + i) * NNH + m] = S[i];
}

// ---------------------------------------------------------------------------
// GroupNorm (groups = heads, 64 ch each) * ln_w + ln_b, * silu(g) -> bf16
// ---------------------------------------------------------------------------
__global__ __launch_bounds__(256) void k_gnorm_gate(
    const float* __restrict__ wkv, const float* __restrict__ gm,
    const float* __restrict__ lnw, const float* __restrict__ lnb,
    unsigned short* __restrict__ y) {
  const int row = blockIdx.x;
  const int tid = threadIdx.x;
  const int c0 = tid * 8;
  const float4* p = (const float4*)&wkv[(size_t)row * CCH + c0];
  float4 a = p[0], bq = p[1];
  float x[8] = {a.x, a.y, a.z, a.w, bq.x, bq.y, bq.z, bq.w};
  float s = 0.f, ss = 0.f;
#pragma unroll
  for (int i = 0; i < 8; ++i) { s += x[i]; ss += x[i] * x[i]; }
  for (int off = 1; off < 8; off <<= 1) {
    s += __shfl_xor(s, off, 8);
    ss += __shfl_xor(ss, off, 8);
  }
  const float mean = s * (1.f / 64.f);
  const float var = ss * (1.f / 64.f) - mean * mean;
  const float inv = rsqrtf(var + EPSV);
  const float4* gp = (const float4*)&gm[(size_t)row * CCH + c0];
  float4 g0 = gp[0], g1 = gp[1];
  float gv[8] = {g0.x, g0.y, g0.z, g0.w, g1.x, g1.y, g1.z, g1.w};
  unsigned int pk[4];
#pragma unroll
  for (int i = 0; i < 4; ++i) {
    float y0 = (x[2 * i] - mean) * inv * lnw[c0 + 2 * i] + lnb[c0 + 2 * i];
    float y1 = (x[2 * i + 1] - mean) * inv * lnw[c0 + 2 * i + 1] + lnb[c0 + 2 * i + 1];
    float gg0 = gv[2 * i], gg1 = gv[2 * i + 1];
    gg0 = gg0 / (1.f + __expf(-gg0));
    gg1 = gg1 / (1.f + __expf(-gg1));
    pk[i] = (unsigned int)f2bf(y0 * gg0) | ((unsigned int)f2bf(y1 * gg1) << 16);
  }
  uint4 o = make_uint4(pk[0], pk[1], pk[2], pk[3]);
  *(uint4*)&y[(size_t)row * CCH + c0] = o;
}

// ---------------------------------------------------------------------------
extern "C" void kernel_launch(void* const* d_in, const int* in_sizes, int n_in,
                              void* d_out, int out_size, void* d_ws,
                              size_t ws_size, hipStream_t stream) {
  (void)in_sizes; (void)n_in; (void)out_size; (void)ws_size;
  const float* hidden = (const float*)d_in[0];
  const float* td = (const float*)d_in[1];
  const float* tf = (const float*)d_in[2];
  const float* Wr = (const float*)d_in[3];
  const float* Wk = (const float*)d_in[4];
  const float* Wv = (const float*)d_in[5];
  const float* Wg = (const float*)d_in[6];
  const float* Wo = (const float*)d_in[7];
  const float* mr = (const float*)d_in[8];
  const float* mk = (const float*)d_in[9];
  const float* mv = (const float*)d_in[10];
  const float* mg = (const float*)d_in[11];
  const float* lnw = (const float*)d_in[12];
  const float* lnb = (const float*)d_in[13];
  const float* init_state = (const float*)d_in[14];

  float* out = (float*)d_out;
  float* fstate = out + (size_t)MMR * CCH;   // final_state after [B,T,C] out

  char* ws = (char*)d_ws;
  const size_t CC2 = (size_t)CCH * CCH * 2;  // bf16 weight bytes
  const size_t MC2 = (size_t)MMR * CCH * 2;  // bf16 activation bytes
  const size_t MC4 = (size_t)MMR * CCH * 4;  // f32 activation bytes

  unsigned short* WrT = (unsigned short*)(ws + 0 * CC2);
  unsigned short* WkT = (unsigned short*)(ws + 1 * CC2);
  unsigned short* WvT = (unsigned short*)(ws + 2 * CC2);
  unsigned short* WgT = (unsigned short*)(ws + 3 * CC2);
  unsigned short* WoT = (unsigned short*)(ws + 4 * CC2);
  char* xbase = ws + 5 * CC2;
  unsigned short* xr = (unsigned short*)(xbase + 0 * MC2);
  unsigned short* xk = (unsigned short*)(xbase + 1 * MC2);
  unsigned short* xv = (unsigned short*)(xbase + 2 * MC2);
  unsigned short* xg = (unsigned short*)(xbase + 3 * MC2);
  char* rbase = xbase + 4 * MC2;
  float* rr = (float*)(rbase + 0 * MC4);
  float* kk = (float*)(rbase + 1 * MC4);
  float* vv = (float*)(rbase + 2 * MC4);
  float* gmat = (float*)(rbase + 3 * MC4);
  // x buffers are dead after the projection GEMMs -> reuse:
  float* wkv = (float*)(xbase);                       // MC4 over xr+xk
  unsigned short* y = (unsigned short*)(xbase + MC4); // MC2 over xv

  dim3 tb(32, 8), tg(CCH / 32, CCH / 32);
  k_transpose_bf16<<<tg, tb, 0, stream>>>(Wr, WrT);
  k_transpose_bf16<<<tg, tb, 0, stream>>>(Wk, WkT);
  k_transpose_bf16<<<tg, tb, 0, stream>>>(Wv, WvT);
  k_transpose_bf16<<<tg, tb, 0, stream>>>(Wg, WgT);
  k_transpose_bf16<<<tg, tb, 0, stream>>>(Wo, WoT);

  k_mix<<<(MMR * (CCH / 4)) / 256, 256, 0, stream>>>(hidden, mr, mk, mv, mg,
                                                     xr, xk, xv, xg);

  dim3 gg(CCH / BN, MMR / BM);
  k_gemm_bf16<<<gg, 256, 0, stream>>>(xr, WrT, rr, CCH, CCH);
  k_gemm_bf16<<<gg, 256, 0, stream>>>(xk, WkT, kk, CCH, CCH);
  k_gemm_bf16<<<gg, 256, 0, stream>>>(xv, WvT, vv, CCH, CCH);
  k_gemm_bf16<<<gg, 256, 0, stream>>>(xg, WgT, gmat, CCH, CCH);

  k_wkv<<<BQ * HH, 256, 0, stream>>>(rr, kk, vv, td, tf, init_state, wkv,
                                     fstate);

  k_gnorm_gate<<<MMR, 256, 0, stream>>>(wkv, gmat, lnw, lnb, y);

  k_gemm_bf16<<<gg, 256, 0, stream>>>(y, WoT, out, CCH, CCH);
}
